// RSSM_27590869909867
// MI455X (gfx1250) — matrix operand order repaired and checked by reading, not compiled, not run on hardware
//
#include <hip/hip_runtime.h>
#include <hip/hip_bf16.h>

#define LSEQ 512
#define BB   256

typedef __attribute__((ext_vector_type(16))) __bf16 v16bf;
typedef __attribute__((ext_vector_type(8)))  float  v8f;

union BFragU { v16bf v; unsigned short h[16]; uint4 q[2]; };

// ---- packed bf16 weight offsets (ushort units) in d_ws ----
constexpr size_t PU_W1 = 0;          // 64x256
constexpr size_t PU_W2 = 16384;      // 256x256
constexpr size_t PX_W1 = 81920;      // 128x256
constexpr size_t PX_W2 = 114688;     // 256x256
constexpr size_t PS_W1 = 180224;     // 64x256
constexpr size_t PS_W2 = 196608;     // 256x256
constexpr size_t PHO_W1 = 262144;    // 512x256
constexpr size_t PHO_W2 = 393216;    // 256x256
constexpr size_t PSA_W1 = 458752;    // 512x256
constexpr size_t PSA_W2 = 589824;    // 256x256
constexpr size_t POST_W1 = 655360;   // 256x512
constexpr size_t POST_W2 = 786432;   // 256x512
constexpr size_t POST_WMU = 917504;  // 512x64
constexpr size_t POST_WLS = 950272;  // 512x64
constexpr size_t GRU_WIH = 983040;   // 256x768
constexpr size_t GRU_WHH = 1179648;  // 256x768
constexpr size_t PACK_TOTAL = 1376256;
constexpr size_t HN_WS_BYTES = PACK_TOTAL * 2; // f32 hn [256,256] after packed weights

// ---- output segment offsets (floats) ----
constexpr size_t OUT_MU   = 0;
constexpr size_t OUT_LS   = 8388608;
constexpr size_t OUT_S    = 16777216;
constexpr size_t OUT_HSEQ = 25165824;
constexpr size_t OUT_XE   = 58720256;
constexpr size_t OUT_HN   = 92274688;

__device__ __forceinline__ unsigned short f2bf(float f) {
  unsigned u = __builtin_bit_cast(unsigned, f);
  u += 0x7FFFu + ((u >> 16) & 1u);           // round-to-nearest-even
  return (unsigned short)(u >> 16);
}

// One 16x16 f32 tile: A = 16 x Kd bf16 (LDS, row-major, stride lda elements),
// B = packed bf16 weight fragments ((nt*kTiles+kt)*512 + lane*16 layout).
// Inner loop: 2x ds_load_b128 (A runs) + 2x global_load_b128 (B frag) + 1 wmma.
__device__ __forceinline__ v8f wmma_tile_bf(const unsigned short* A, int lda,
                                            const unsigned short* Wp, int Kd,
                                            int nt, int lane, v8f acc) {
  const int m = lane & 15;
  const int hi = lane >> 4;
  const int kTiles = Kd >> 5;
  const unsigned short* arow = A + (size_t)m * lda;
  const unsigned short* wbase = Wp + ((size_t)nt * kTiles) * 512 + (size_t)lane * 16;
  for (int kt = 0; kt < kTiles; ++kt) {
    const int k0 = kt << 5;
    BFragU af;
    af.q[0] = *(const uint4*)(arow + k0 + hi * 8);        // K = k0 + hi*8 .. +7
    af.q[1] = *(const uint4*)(arow + k0 + 16 + hi * 8);   // K = k0+16+hi*8 .. +7
    BFragU bf;
    const uint4* wq = (const uint4*)(wbase + (size_t)kt * 512);
    bf.q[0] = wq[0]; bf.q[1] = wq[1];
    acc = __builtin_amdgcn_wmma_f32_16x16x32_bf16(false, af.v, false, bf.v,
                                                  (short)0, acc, false, false);
  }
  return acc;
}

// epilogue: bias + relu -> bf16 LDS (A-ready for the next GEMM)
__device__ __forceinline__ void tile_store_relu_bf(unsigned short* O, int ldo, int lane,
                                                   v8f acc, const float* bias, int n0) {
  const int col = lane & 15;
  const int rb = (lane >> 4) * 8;
  const float b = bias[n0 + col];
#pragma unroll
  for (int r = 0; r < 8; ++r) {
    float v = acc[r] + b;
    O[(size_t)(rb + r) * ldo + col] = f2bf(v > 0.f ? v : 0.f);
  }
}

// epilogue: bias + relu -> f32 global (final x_emb output)
__device__ __forceinline__ void tile_store_relu_f32(float* O, int ldo, int lane,
                                                    v8f acc, const float* bias, int n0) {
  const int col = lane & 15;
  const int rb = (lane >> 4) * 8;
  const float b = bias[n0 + col];
#pragma unroll
  for (int r = 0; r < 8; ++r) {
    float v = acc[r] + b;
    O[(size_t)(rb + r) * ldo + col] = v > 0.f ? v : 0.f;
  }
}

__device__ __forceinline__ float sigmoidf_(float x) { return 1.f / (1.f + __expf(-x)); }

// ---- weight pack: f32 [Kd,N] row-major -> bf16 WMMA-B fragments ----
__global__ void __launch_bounds__(256) packw_kernel(const float* __restrict__ W, int Kd, int N,
                                                    unsigned short* __restrict__ Wp, int total) {
  int e = blockIdx.x * 256 + threadIdx.x;
  if (e >= total) return;
  int frag = e >> 9, within = e & 511;
  int lane = within >> 4, h = within & 15;
  int kTiles = Kd >> 5;
  int nt = frag / kTiles, kt = frag - nt * kTiles;
  int k = kt * 32 + ((lane >> 4) << 3) + (h & 7) + ((h >> 3) << 4);   // lanes 16..31 hold K = k0+8+j (j<8), k0+24+(j-8) (j>=8)
  int n = nt * 16 + (lane & 15);
  Wp[e] = f2bf(W[(size_t)k * N + n]);
}

__global__ void __launch_bounds__(256) zero_kernel(float* p, int n) {
  int i = blockIdx.x * 256 + threadIdx.x;
  if (i < n) p[i] = 0.f;
}

// ---- fused 2-layer ReLU MLP over 16-row blocks (x_emb prepass, f32 out) ----
__global__ void __launch_bounds__(256) embed2_kernel(const float* __restrict__ X, int Din,
                                                     const unsigned short* __restrict__ W1, const float* __restrict__ b1,
                                                     const unsigned short* __restrict__ W2, const float* __restrict__ b2,
                                                     float* __restrict__ Out) {
  __shared__ __align__(16) unsigned short sx[16 * 128];
  __shared__ __align__(16) unsigned short sm[16 * 256];
  const int tid = threadIdx.x, lane = tid & 31, wid = tid >> 5;
  const size_t row0 = (size_t)blockIdx.x * 16;
  for (int i = tid; i < 16 * Din; i += 256)
    sx[i] = f2bf(X[row0 * Din + i]);
  __syncthreads();
  for (int nt = wid; nt < 16; nt += 8) {
    v8f acc = {};
    acc = wmma_tile_bf(sx, Din, W1, Din, nt, lane, acc);
    tile_store_relu_bf(sm + nt * 16, 256, lane, acc, b1, nt * 16);
  }
  __syncthreads();
  float* O = Out + row0 * 256;
  for (int nt = wid; nt < 16; nt += 8) {
    v8f acc = {};
    acc = wmma_tile_bf(sm, 256, W2, 256, nt, lane, acc);
    tile_store_relu_f32(O + nt * 16, 256, lane, acc, b2, nt * 16);
  }
}

struct StepParams {
  const float *u_in, *eps, *xe;
  float *hn, *out;
  const unsigned short* P;
  const float *pu_b1, *pu_b2, *pho_b1, *pho_b2, *ps_b1, *ps_b2, *psa_b1, *psa_b2;
  const float *post_b1, *post_b2, *post_bmu, *post_bls, *gru_bih, *gru_bhh;
};

// One timestep: each WG owns 16 batch rows end-to-end. grid = B/16 = 16 WGs.
__global__ void __launch_bounds__(256) rssm_step_kernel(StepParams p, int t) {
  __shared__ __align__(16) unsigned short sb[25600];  // 50 KB bf16 activations
  __shared__ __align__(16) float sf[2048];            // 8 KB mu/ls f32
  unsigned short* CC  = sb;          // [16][512] concat inputs
  unsigned short* MID = sb + 8192;   // [16][512] hidden / tprod (+u stage at +4096)
  unsigned short* ACT = sb + 16384;  // [16][256] hid / hid2
  unsigned short* UE  = sb + 20480;  // [16][256] u_emb, later hn snapshot
  unsigned short* SBF = sb + 24576;  // [16][64] sampled state bf16
  const int tid = threadIdx.x, lane = tid & 31, wid = tid >> 5;
  const int row0 = blockIdx.x * 16;

  // stage concat [hn | x_emb[t]] -> CC (f32 -> bf16)
  for (int i = tid; i < 16 * 256; i += 256) {
    int r = i >> 8, c = i & 255;
    CC[r * 512 + c]       = f2bf(p.hn[(size_t)(row0 + r) * 256 + c]);
    CC[r * 512 + 256 + c] = f2bf(p.xe[((size_t)t * 256 + row0 + r) * 256 + c]);
  }
  __syncthreads();
  // S1a: pho layer1 (512->256)
  for (int nt = wid; nt < 16; nt += 8) {
    v8f acc = {}; acc = wmma_tile_bf(CC, 512, p.P + PHO_W1, 512, nt, lane, acc);
    tile_store_relu_bf(MID + nt * 16, 256, lane, acc, p.pho_b1, nt * 16);
  }
  __syncthreads();
  // S1b: pho layer2 (256->256) -> hid in ACT
  for (int nt = wid; nt < 16; nt += 8) {
    v8f acc = {}; acc = wmma_tile_bf(MID, 256, p.P + PHO_W2, 256, nt, lane, acc);
    tile_store_relu_bf(ACT + nt * 16, 256, lane, acc, p.pho_b2, nt * 16);
  }
  __syncthreads();
  // S2a: tprod = tanh(hid@W1+b1)*sigmoid(hid@W2+b2) -> MID [16][512] bf16
  for (int nt = wid; nt < 32; nt += 8) {
    v8f a1 = {}; a1 = wmma_tile_bf(ACT, 256, p.P + POST_W1, 256, nt, lane, a1);
    v8f a2 = {}; a2 = wmma_tile_bf(ACT, 256, p.P + POST_W2, 256, nt, lane, a2);
    int col = lane & 15, rb = (lane >> 4) * 8;
    float b1v = p.post_b1[nt * 16 + col], b2v = p.post_b2[nt * 16 + col];
#pragma unroll
    for (int r = 0; r < 8; ++r)
      MID[(size_t)(rb + r) * 512 + nt * 16 + col] =
          f2bf(tanhf(a1[r] + b1v) * sigmoidf_(a2[r] + b2v));
  }
  __syncthreads();
  // S2b: mu (waves 0-3), logsigma (waves 4-7): [16,512]@[512,64] -> sf (f32)
  {
    int sub = wid & 3;
    const unsigned short* Wm = (wid < 4) ? (p.P + POST_WMU) : (p.P + POST_WLS);
    const float* bm = (wid < 4) ? p.post_bmu : p.post_bls;
    float* Om = (wid < 4) ? sf : (sf + 1024);
    v8f acc = {}; acc = wmma_tile_bf(MID, 512, Wm, 512, sub, lane, acc);
    int col = lane & 15, rb = (lane >> 4) * 8;
    float bv = bm[sub * 16 + col];
#pragma unroll
    for (int r = 0; r < 8; ++r) Om[(rb + r) * 64 + sub * 16 + col] = acc[r] + bv;
  }
  __syncthreads();
  // S2c: s = mu + exp(ls)*eps ; emit mu/ls/s f32; s -> SBF bf16; stage u[t] -> MID+4096
  for (int i = tid; i < 16 * 64; i += 256) {
    int r = i >> 6, c = i & 63;
    size_t oi = ((size_t)t * 256 + row0 + r) * 64 + c;
    float mu = sf[i], ls = sf[1024 + i];
    float s = mu + __expf(ls) * p.eps[oi];
    p.out[OUT_MU + oi] = mu;
    p.out[OUT_LS + oi] = ls;
    p.out[OUT_S + oi]  = s;
    SBF[i] = f2bf(s);
    MID[4096 + i] = f2bf(p.u_in[oi]);   // u[t] rows, [16][64] bf16
  }
  __syncthreads();
  // S0a: u_emb layer1 (64->256), A staged at MID+4096 (disjoint from output region)
  for (int nt = wid; nt < 16; nt += 8) {
    v8f acc = {}; acc = wmma_tile_bf(MID + 4096, 64, p.P + PU_W1, 64, nt, lane, acc);
    tile_store_relu_bf(MID + nt * 16, 256, lane, acc, p.pu_b1, nt * 16);
  }
  __syncthreads();
  // S0b: u_emb layer2 -> UE
  for (int nt = wid; nt < 16; nt += 8) {
    v8f acc = {}; acc = wmma_tile_bf(MID, 256, p.P + PU_W2, 256, nt, lane, acc);
    tile_store_relu_bf(UE + nt * 16, 256, lane, acc, p.pu_b2, nt * 16);
  }
  __syncthreads();
  // S3a: se layer1 (64->256) from s
  for (int nt = wid; nt < 16; nt += 8) {
    v8f acc = {}; acc = wmma_tile_bf(SBF, 64, p.P + PS_W1, 64, nt, lane, acc);
    tile_store_relu_bf(MID + nt * 16, 256, lane, acc, p.ps_b1, nt * 16);
  }
  __syncthreads();
  // S3b: se layer2 -> CC[:, :256]; copy UE -> CC[:, 256:]
  for (int nt = wid; nt < 16; nt += 8) {
    v8f acc = {}; acc = wmma_tile_bf(MID, 256, p.P + PS_W2, 256, nt, lane, acc);
    tile_store_relu_bf(CC + nt * 16, 512, lane, acc, p.ps_b2, nt * 16);
  }
  for (int i = tid; i < 16 * 256; i += 256) {
    int r = i >> 8, c = i & 255;
    CC[r * 512 + 256 + c] = UE[r * 256 + c];
  }
  __syncthreads();
  // S4a: psa layer1 (512->256); snapshot hn -> UE bf16 (UE free now)
  for (int nt = wid; nt < 16; nt += 8) {
    v8f acc = {}; acc = wmma_tile_bf(CC, 512, p.P + PSA_W1, 512, nt, lane, acc);
    tile_store_relu_bf(MID + nt * 16, 256, lane, acc, p.psa_b1, nt * 16);
  }
  for (int i = tid; i < 16 * 256; i += 256) {
    int r = i >> 8, c = i & 255;
    UE[r * 256 + c] = f2bf(p.hn[(size_t)(row0 + r) * 256 + c]);
  }
  __syncthreads();
  // S4b: psa layer2 -> hid2 in ACT; h_seq[t] = hn (pre-update, exact f32 from global)
  for (int nt = wid; nt < 16; nt += 8) {
    v8f acc = {}; acc = wmma_tile_bf(MID, 256, p.P + PSA_W2, 256, nt, lane, acc);
    tile_store_relu_bf(ACT + nt * 16, 256, lane, acc, p.psa_b2, nt * 16);
  }
  for (int i = tid; i < 16 * 256; i += 256) {
    int r = i >> 8, c = i & 255;
    p.out[OUT_HSEQ + ((size_t)t * 256 + row0 + r) * 256 + c] =
        p.hn[(size_t)(row0 + r) * 256 + c];
  }
  __syncthreads();
  // S5: GRU — each wave owns 16-column groups; r/z/n tiles kept in registers.
  // gh A comes from the UE snapshot, so concurrent p.hn writes are race-free
  // (each (row,col) of p.hn is read/written by exactly one lane).
  for (int cg = wid; cg < 16; cg += 8) {
    v8f gir = {}, giz = {}, gin = {}, ghr = {}, ghz = {}, ghn = {};
    gir = wmma_tile_bf(ACT, 256, p.P + GRU_WIH, 256, cg,      lane, gir);
    giz = wmma_tile_bf(ACT, 256, p.P + GRU_WIH, 256, 16 + cg, lane, giz);
    gin = wmma_tile_bf(ACT, 256, p.P + GRU_WIH, 256, 32 + cg, lane, gin);
    ghr = wmma_tile_bf(UE, 256, p.P + GRU_WHH, 256, cg,      lane, ghr);
    ghz = wmma_tile_bf(UE, 256, p.P + GRU_WHH, 256, 16 + cg, lane, ghz);
    ghn = wmma_tile_bf(UE, 256, p.P + GRU_WHH, 256, 32 + cg, lane, ghn);
    int col = cg * 16 + (lane & 15), rb = (lane >> 4) * 8;
    float bir = p.gru_bih[col], biz = p.gru_bih[256 + col], bin = p.gru_bih[512 + col];
    float bhr = p.gru_bhh[col], bhz = p.gru_bhh[256 + col], bhn = p.gru_bhh[512 + col];
#pragma unroll
    for (int r = 0; r < 8; ++r) {
      float rr = sigmoidf_(gir[r] + bir + ghr[r] + bhr);
      float zz = sigmoidf_(giz[r] + biz + ghz[r] + bhz);
      float nn = tanhf(gin[r] + bin + rr * (ghn[r] + bhn));
      size_t hi_ = (size_t)(row0 + rb + r) * 256 + col;
      float hold = p.hn[hi_];
      float hnew = (1.f - zz) * nn + zz * hold;
      p.hn[hi_] = hnew;
      if (t == LSEQ - 1) p.out[OUT_HN + hi_] = hnew;
    }
  }
}

extern "C" void kernel_launch(void* const* d_in, const int* in_sizes, int n_in,
                              void* d_out, int out_size, void* d_ws, size_t ws_size,
                              hipStream_t stream) {
  (void)in_sizes; (void)n_in; (void)out_size; (void)ws_size;
  const float* u_in = (const float*)d_in[0];
  const float* obs  = (const float*)d_in[1];
  const float* eps  = (const float*)d_in[2];
  unsigned short* P = (unsigned short*)d_ws;
  float* hn  = (float*)((char*)d_ws + HN_WS_BYTES);
  float* out = (float*)d_out;

  struct PackSpec { const float* w; int Kd, N; size_t off; };
  const PackSpec specs[16] = {
    {(const float*)d_in[3],  64,  256, PU_W1},  {(const float*)d_in[5],  256, 256, PU_W2},
    {(const float*)d_in[7],  128, 256, PX_W1},  {(const float*)d_in[9],  256, 256, PX_W2},
    {(const float*)d_in[11], 64,  256, PS_W1},  {(const float*)d_in[13], 256, 256, PS_W2},
    {(const float*)d_in[15], 512, 256, PHO_W1}, {(const float*)d_in[17], 256, 256, PHO_W2},
    {(const float*)d_in[19], 512, 256, PSA_W1}, {(const float*)d_in[21], 256, 256, PSA_W2},
    {(const float*)d_in[23], 256, 512, POST_W1},{(const float*)d_in[25], 256, 512, POST_W2},
    {(const float*)d_in[27], 512, 64,  POST_WMU},{(const float*)d_in[29], 512, 64, POST_WLS},
    {(const float*)d_in[31], 256, 768, GRU_WIH},{(const float*)d_in[32], 256, 768, GRU_WHH},
  };
  for (int i = 0; i < 16; ++i) {
    int total = specs[i].Kd * specs[i].N;
    packw_kernel<<<(total + 255) / 256, 256, 0, stream>>>(specs[i].w, specs[i].Kd,
                                                          specs[i].N, P + specs[i].off, total);
  }
  zero_kernel<<<(BB * 256 + 255) / 256, 256, 0, stream>>>(hn, BB * 256);
  // x_emb for all L*B rows, straight into d_out
  embed2_kernel<<<(LSEQ * BB) / 16, 256, 0, stream>>>(obs, 128, P + PX_W1,
      (const float*)d_in[8], P + PX_W2, (const float*)d_in[10], out + OUT_XE);

  StepParams sp;
  sp.u_in = u_in; sp.eps = eps; sp.xe = out + OUT_XE; sp.hn = hn; sp.out = out; sp.P = P;
  sp.pu_b1 = (const float*)d_in[4];   sp.pu_b2 = (const float*)d_in[6];
  sp.pho_b1 = (const float*)d_in[16]; sp.pho_b2 = (const float*)d_in[18];
  sp.ps_b1 = (const float*)d_in[12];  sp.ps_b2 = (const float*)d_in[14];
  sp.psa_b1 = (const float*)d_in[20]; sp.psa_b2 = (const float*)d_in[22];
  sp.post_b1 = (const float*)d_in[24]; sp.post_b2 = (const float*)d_in[26];
  sp.post_bmu = (const float*)d_in[28]; sp.post_bls = (const float*)d_in[30];
  sp.gru_bih = (const float*)d_in[33]; sp.gru_bhh = (const float*)d_in[34];
  for (int t = 0; t < LSEQ; ++t)
    rssm_step_kernel<<<BB / 16, 256, 0, stream>>>(sp, t);
}
